// MetaLayer_13331578487489
// MI455X (gfx1250) — matrix operand order repaired and checked by reading, not compiled, not run on hardware
//
#include <hip/hip_runtime.h>
#include <cstdint>
#include <cstddef>

typedef unsigned short u16;
typedef __attribute__((ext_vector_type(16))) __bf16 v16bf;
typedef __attribute__((ext_vector_type(8)))  float  v8f;

#define DEV static __device__ __forceinline__

DEV u16 f2bf(float f) {                      // f32 -> bf16, round to nearest even
  unsigned u = __float_as_uint(f);
  u += 0x7FFFu + ((u >> 16) & 1u);
  return (u16)(u >> 16);
}
DEV float bf2f(u16 h) { return __uint_as_float(((unsigned)h) << 16); }
DEV float sigm(float a) { return 1.f / (1.f + __expf(-a)); }
DEV float silu(float a) { return a * sigm(a); }

union FragBf { uint4 q[2]; v16bf v; };

// CDNA5 async global->LDS DMA (ASYNCcnt-tracked, no VGPR staging).
// LDS operand is the wave-relative LDS byte offset = low 32 bits of the flat
// pointer (ISA 10.2: LDS aperture addresses are {SHARED_BASE, offset[31:0]}).
DEV void async_ld_b128(const u16* gsrc, u16* ldst) {
  asm volatile("global_load_async_to_lds_b128 %0, %1, off"
               :: "v"((unsigned)(uintptr_t)(void*)ldst), "v"(gsrc)
               : "memory");
}
#if __has_builtin(__builtin_amdgcn_s_wait_asynccnt)
#define WAIT_ASYNC(n) __builtin_amdgcn_s_wait_asynccnt(n)
#else
#define WAIT_ASYNC(n) asm volatile("s_wait_asynccnt %0" :: "i"(n) : "memory")
#endif

// ---------------------------------------------------------------------------
// Batched bf16 WMMA GEMM (TN): C[M,N](f32 or bf16) = A[M,K] @ Bt[N,K]^T + bias
//   A : row-major [M,K] bf16 (K contiguous)
//   Bt: row-major [N,K] bf16 (K contiguous)
// Block tile 128x128x32; 256 threads = 8 wave32 (2x4), 64x32 per wave.
// Global->LDS via async DMA, 3-deep LDS pipeline. M,N mult of 128, K of 32.
// ---------------------------------------------------------------------------
template<bool OUT_BF16>
__global__ __launch_bounds__(256) void gemm_wmma(
    const u16* __restrict__ A,  int lda, long long sA,
    const u16* __restrict__ Bt, int ldb, long long sB,
    void* __restrict__ Cp,      int ldc, long long sC,
    const float* __restrict__ bias, long long sBias,
    int K)
{
  constexpr int BKt = 32, LSTR = 40;          // 40-ushort LDS row stride
  __shared__ u16 As[3][128 * LSTR];
  __shared__ u16 Bs[3][128 * LSTR];

  const int tid = threadIdx.x, lane = tid & 31, wave = tid >> 5;
  const int wm = (wave & 1) * 64, wn = (wave >> 1) * 32;
  const long long bM = (long long)blockIdx.y * 128;
  const long long bN = (long long)blockIdx.x * 128;
  const u16* Ab = A  + (long long)blockIdx.z * sA;
  const u16* Bb = Bt + (long long)blockIdx.z * sB;

  // This thread's two 16-byte chunks of each 128x32 tile: chunk c -> row c>>2,
  // 8-elem group c&3. Each async op moves 16B/lane (b128).
  const int r0 = tid >> 2,          q0 = (tid & 3) * 8;
  const int r1 = (tid + 256) >> 2,  q1 = ((tid + 256) & 3) * 8;

  auto issue = [&](int kt, int buf) {
    const long long ko = (long long)kt * BKt;
    async_ld_b128(Ab + (bM + r0) * lda + ko + q0, &As[buf][r0 * LSTR + q0]);
    async_ld_b128(Ab + (bM + r1) * lda + ko + q1, &As[buf][r1 * LSTR + q1]);
    async_ld_b128(Bb + (bN + r0) * ldb + ko + q0, &Bs[buf][r0 * LSTR + q0]);
    async_ld_b128(Bb + (bN + r1) * ldb + ko + q1, &Bs[buf][r1 * LSTR + q1]);
  };

  v8f acc[4][2] = {};
  const int nkt = K / BKt;
  issue(0, 0);
  if (nkt > 1) issue(1, 1);

  for (int kt = 0; kt < nkt; ++kt) {
    // tile kt complete (tile kt+1 may remain in flight: 4 ops/wave)
    if (kt + 1 < nkt) WAIT_ASYNC(4); else WAIT_ASYNC(0);
    __syncthreads();
    if (kt + 2 < nkt) issue(kt + 2, (kt + 2) % 3);

    const int cur = kt % 3;
    // A frags: lanes 0-15 hold M=lane, K 0..7 & 16..23; lanes 16-31 K 8..15 & 24..31
    FragBf fa[4], fb[2];
    const int arow = wm + (lane & 15), ahalf = (lane >> 4) * 8;
#pragma unroll
    for (int i = 0; i < 4; ++i) {
      const u16* p = &As[cur][(arow + 16 * i) * LSTR + ahalf];
      fa[i].q[0] = *(const uint4*)p;
      fa[i].q[1] = *(const uint4*)(p + 16);
    }
    // B frags: lanes 0-15 hold N=lane, K 0..7 & 16..23; lanes 16-31 K 8..15 & 24..31
    const int bcol = wn + (lane & 15), bhalf = (lane >> 4) * 8;
#pragma unroll
    for (int j = 0; j < 2; ++j) {
      const u16* p = &Bs[cur][(bcol + 16 * j) * LSTR + bhalf];
      fb[j].q[0] = *(const uint4*)p;
      fb[j].q[1] = *(const uint4*)(p + 16);
    }
#pragma unroll
    for (int i = 0; i < 4; ++i)
#pragma unroll
      for (int j = 0; j < 2; ++j)
        acc[i][j] = __builtin_amdgcn_wmma_f32_16x16x32_bf16(
            false, fa[i].v, false, fb[j].v, (short)0, acc[i][j], false, false);
  }

  // Epilogue: C/D layout — lane<16: N=lane,M=vr; lane>=16: N=lane-16,M=vr+8
  const long long cbase = (long long)blockIdx.z * sC;
  const float* bp = bias ? bias + (long long)blockIdx.z * sBias : nullptr;
  const int crow0 = (int)bM + wm + (lane >> 4) * 8;
  const int ccol0 = (int)bN + wn + (lane & 15);
#pragma unroll
  for (int i = 0; i < 4; ++i)
#pragma unroll
    for (int j = 0; j < 2; ++j) {
      const int col = ccol0 + 16 * j;
      const float bv = bp ? bp[col] : 0.f;
#pragma unroll
      for (int r = 0; r < 8; ++r) {
        const long long idx = cbase + (long long)(crow0 + 16 * i + r) * ldc + col;
        const float v = acc[i][j][r] + bv;
        if (OUT_BF16) ((u16*)Cp)[idx] = f2bf(v);
        else          ((float*)Cp)[idx] = v;
      }
    }
}

// ---------------------------------------------------------------------------
// Elementwise / helper kernels
// ---------------------------------------------------------------------------
__global__ void k_cvt(const float* __restrict__ in, u16* __restrict__ out, long long n) {
  long long i = (long long)blockIdx.x * blockDim.x + threadIdx.x;
  if (i < n) out[i] = f2bf(in[i]);
}

template<typename T>   // in [b,R,C] (f32 or bf16) -> out [b,C,R] bf16
__global__ void k_transpose(const T* __restrict__ in, long long sIn,
                            u16* __restrict__ out, long long sOut, int R, int C) {
  __shared__ u16 tile[32][33];
  const T* ib = in + (long long)blockIdx.z * sIn;
  u16* ob = out + (long long)blockIdx.z * sOut;
  const int c0 = blockIdx.x * 32, r0 = blockIdx.y * 32;
  const int tx = threadIdx.x, ty = threadIdx.y;
#pragma unroll
  for (int i = 0; i < 4; ++i) {
    const int r = r0 + ty + i * 8;
    if constexpr (sizeof(T) == 4) tile[ty + i * 8][tx] = f2bf((float)ib[(long long)r * C + c0 + tx]);
    else                          tile[ty + i * 8][tx] = (u16)ib[(long long)r * C + c0 + tx];
  }
  __syncthreads();
#pragma unroll
  for (int i = 0; i < 4; ++i)
    ob[(long long)(c0 + ty + i * 8) * R + r0 + tx] = tile[tx][ty + i * 8];
}

__global__ void k_swiglu(const u16* __restrict__ H, u16* __restrict__ S, int F, long long total) {
  long long i = (long long)blockIdx.x * blockDim.x + threadIdx.x;
  if (i >= total) return;
  const long long m = i / F; const int j = (int)(i % F);
  const float a = bf2f(H[m * (2LL * F) + j]);
  const float g = bf2f(H[m * (2LL * F) + F + j]);
  S[i] = f2bf(silu(a) * g);
}

__global__ void k_dswiglu(const u16* __restrict__ dS, const u16* __restrict__ H,
                          u16* __restrict__ dH, int F, long long total) {
  long long i = (long long)blockIdx.x * blockDim.x + threadIdx.x;
  if (i >= total) return;
  const long long m = i / F; const int j = (int)(i % F);
  const float a = bf2f(H[m * (2LL * F) + j]);
  const float g = bf2f(H[m * (2LL * F) + F + j]);
  const float sg = sigm(a);
  const float ds = bf2f(dS[i]);
  dH[m * (2LL * F) + j]     = f2bf(ds * g * (sg * (1.f + a * (1.f - sg))));
  dH[m * (2LL * F) + F + j] = f2bf(ds * (a * sg));
}

// cosine-distance loss gradient; one block per token row, 16 groups of 64
__global__ __launch_bounds__(256) void k_lossgrad(const float* __restrict__ pr,
    const float* __restrict__ tg, u16* __restrict__ dpr) {
  const int row = blockIdx.x, t = threadIdx.x;
  const long long base = (long long)row * 1024;
  __shared__ float pn[256], pp[256], pt[256];
  __shared__ float cA[16], cB[16];
  float n0 = 0, p0 = 0, t0 = 0, pv[4], tv[4];
#pragma unroll
  for (int e = 0; e < 4; ++e) {
    const int c = t * 4 + e;
    pv[e] = pr[base + c]; tv[e] = tg[base + c];
    n0 += pv[e] * tv[e]; p0 += pv[e] * pv[e]; t0 += tv[e] * tv[e];
  }
  pn[t] = n0; pp[t] = p0; pt[t] = t0;
  __syncthreads();
  if (t < 16) {
    float num = 0, p2 = 0, t2 = 0;
    for (int k = 0; k < 16; ++k) { num += pn[t * 16 + k]; p2 += pp[t * 16 + k]; t2 += pt[t * 16 + k]; }
    const float den = fmaxf(sqrtf(p2) * sqrtf(t2), 1e-8f);
    const float invN = 1.f / 2048.f;
    cA[t] =  invN * num / (fmaxf(p2, 1e-20f) * den);
    cB[t] = -invN / den;
  }
  __syncthreads();
  const int grp = t >> 4;
#pragma unroll
  for (int e = 0; e < 4; ++e) {
    const int c = t * 4 + e;
    dpr[base + c] = f2bf(cA[grp] * pv[e] + cB[grp] * tv[e]);
  }
}

__global__ void k_colsum(const u16* __restrict__ X, float* __restrict__ out,
                         int rows, int cols) {
  const int c = blockIdx.x * blockDim.x + threadIdx.x;
  const int b = blockIdx.y;
  const u16* Xb = X + (long long)b * rows * cols;
  float a = 0;
  for (int r = 0; r < rows; ++r) a += bf2f(Xb[(long long)r * cols + c]);
  out[(long long)b * cols + c] = a;
}

// Wt[b][n,k] = bf16(uW[k,n] - |lr| * gW[b,k,n])   (fused fast-weight + transpose)
__global__ void k_fastw(const float* __restrict__ uW, const float* __restrict__ gW,
                        const float* __restrict__ lrp, u16* __restrict__ Wt,
                        int K, int Nc) {
  const float la = fabsf(lrp[0]);
  __shared__ u16 tile[32][33];
  const int b = blockIdx.z;
  const int n0 = blockIdx.x * 32, k0 = blockIdx.y * 32;
  const int tx = threadIdx.x, ty = threadIdx.y;
#pragma unroll
  for (int i = 0; i < 4; ++i) {
    const int k = k0 + ty + i * 8;
    const long long o = (long long)k * Nc + n0 + tx;
    tile[ty + i * 8][tx] = f2bf(uW[o] - la * gW[(long long)b * K * Nc + o]);
  }
  __syncthreads();
#pragma unroll
  for (int i = 0; i < 4; ++i)
    Wt[((long long)b * Nc + (n0 + ty + i * 8)) * K + k0 + tx] = tile[tx][ty + i * 8];
}

__global__ void k_fastb(const float* __restrict__ ub, const float* __restrict__ gb,
                        const float* __restrict__ lrp, float* __restrict__ bp,
                        int n, long long total) {
  long long i = (long long)blockIdx.x * blockDim.x + threadIdx.x;
  if (i >= total) return;
  bp[i] = ub[i % n] - fabsf(lrp[0]) * gb[i];
}

__global__ __launch_bounds__(256) void k_layernorm(const float* __restrict__ y,
    const float* __restrict__ ga, const float* __restrict__ be, float* __restrict__ out) {
  const int row = blockIdx.x, t = threadIdx.x;
  const float* yr = y + (long long)row * 1024;
  __shared__ float s1[256], s2[256];
  float a = 0, q = 0, v[4];
#pragma unroll
  for (int e = 0; e < 4; ++e) { v[e] = yr[t + e * 256]; a += v[e]; q += v[e] * v[e]; }
  s1[t] = a; s2[t] = q; __syncthreads();
  for (int s = 128; s > 0; s >>= 1) {
    if (t < s) { s1[t] += s1[t + s]; s2[t] += s2[t + s]; }
    __syncthreads();
  }
  const float mu = s1[0] * (1.f / 1024.f);
  const float var = s2[0] * (1.f / 1024.f) - mu * mu;
  const float inv = rsqrtf(var + 1e-5f);
  float* orow = out + (long long)row * 1024;
#pragma unroll
  for (int e = 0; e < 4; ++e) { const int c = t + e * 256; orow[c] = (v[e] - mu) * inv * ga[c] + be[c]; }
}

// ---------------------------------------------------------------------------
extern "C" void kernel_launch(void* const* d_in, const int* in_sizes, int n_in,
                              void* d_out, int out_size, void* d_ws, size_t ws_size,
                              hipStream_t stream) {
  (void)in_sizes; (void)n_in; (void)out_size; (void)ws_size;
  const float* x    = (const float*)d_in[0];
  const float* tg   = (const float*)d_in[1];
  const float* inW1 = (const float*)d_in[2];
  const float* inb1 = (const float*)d_in[3];
  const float* inW2 = (const float*)d_in[4];
  const float* inb2 = (const float*)d_in[5];
  const float* prW1 = (const float*)d_in[6];
  const float* prb1 = (const float*)d_in[7];
  const float* prW2 = (const float*)d_in[8];
  const float* prb2 = (const float*)d_in[9];
  const float* uW1  = (const float*)d_in[10];
  const float* ub1  = (const float*)d_in[11];
  const float* uW2  = (const float*)d_in[12];
  const float* ub2  = (const float*)d_in[13];
  const float* lr   = (const float*)d_in[14];
  const float* lng  = (const float*)d_in[15];
  const float* lnb  = (const float*)d_in[16];

  const int Bd = 8, Nd = 2048;
  const long long MT = (long long)Bd * Nd;                 // 16384 token rows

  char* wsp = (char*)d_ws; size_t off = 0;
  auto alloc = [&](long long bytes) -> char* {
    char* p = wsp + off; off = (off + (size_t)bytes + 255) & ~(size_t)255; return p;
  };

  u16* x_bf   = (u16*)alloc(MT * 1024 * 2);
  u16* xT     = (u16*)alloc((long long)Bd * 1024 * 2048 * 2);
  u16* W1t    = (u16*)alloc(4096LL * 1024 * 2);
  u16* W2t    = (u16*)alloc(1024LL * 2048 * 2);
  u16* W2p    = (u16*)alloc(2048LL * 1024 * 2);
  u16* pW1p   = (u16*)alloc(1024LL * 2048 * 2);
  u16* pW1t   = (u16*)alloc(2048LL * 1024 * 2);
  u16* pW2p   = (u16*)alloc(1024LL * 1024 * 2);
  u16* pW2t   = (u16*)alloc(1024LL * 1024 * 2);
  u16* Hb     = (u16*)alloc(MT * 4096 * 2);                // reused for fast-weight H
  u16* sb     = (u16*)alloc(MT * 2048 * 2);                // reused for fast-weight s
  u16* hb     = (u16*)alloc(MT * 1024 * 2);
  u16* Hpb    = (u16*)alloc(MT * 2048 * 2);
  u16* s2b    = (u16*)alloc(MT * 1024 * 2);
  float* prf  = (float*)alloc(MT * 1024 * 4);              // reused for final y
  u16* dprb   = (u16*)alloc(MT * 1024 * 2);
  u16* ds2b   = (u16*)alloc(MT * 1024 * 2);
  u16* dHpb   = (u16*)alloc(MT * 2048 * 2);
  u16* dhb    = (u16*)alloc(MT * 1024 * 2);
  u16* dsb    = (u16*)alloc(MT * 2048 * 2);
  u16* dHb    = (u16*)alloc(MT * 4096 * 2);
  u16* sTb    = (u16*)alloc((long long)Bd * 2048 * 2048 * 2);
  u16* dHbT   = (u16*)alloc((long long)Bd * 4096 * 2048 * 2);  // dH^T per sample
  u16* dhbT   = (u16*)alloc((long long)Bd * 1024 * 2048 * 2);  // dh^T per sample
  float* gW1  = (float*)alloc((long long)Bd * 1024 * 4096 * 4);
  float* gW2  = (float*)alloc((long long)Bd * 2048 * 1024 * 4);
  float* gb1  = (float*)alloc((long long)Bd * 4096 * 4);
  float* gb2  = (float*)alloc((long long)Bd * 1024 * 4);
  u16* W1pt   = (u16*)alloc((long long)Bd * 4096 * 1024 * 2);
  u16* W2pt   = (u16*)alloc((long long)Bd * 1024 * 2048 * 2);
  float* b1p  = (float*)alloc((long long)Bd * 4096 * 4);
  float* b2p  = (float*)alloc((long long)Bd * 1024 * 4);

  const dim3 tB(256), trB(32, 8);
  auto ew = [](long long n) { return dim3((unsigned)((n + 255) / 256)); };

  // --- convert / transpose operands to bf16 ---
  k_cvt<<<ew(MT * 1024), tB, 0, stream>>>(x, x_bf, MT * 1024);
  k_transpose<float><<<dim3(1024/32, 2048/32, Bd), trB, 0, stream>>>(x, (long long)Nd*1024, xT, 1024LL*2048, Nd, 1024);
  k_transpose<float><<<dim3(4096/32, 1024/32, 1), trB, 0, stream>>>(inW1, 0, W1t, 0, 1024, 4096);
  k_transpose<float><<<dim3(1024/32, 2048/32, 1), trB, 0, stream>>>(inW2, 0, W2t, 0, 2048, 1024);
  k_transpose<float><<<dim3(2048/32, 1024/32, 1), trB, 0, stream>>>(prW1, 0, pW1t, 0, 1024, 2048);
  k_transpose<float><<<dim3(1024/32, 1024/32, 1), trB, 0, stream>>>(prW2, 0, pW2t, 0, 1024, 1024);
  k_cvt<<<ew(2048LL*1024), tB, 0, stream>>>(inW2, W2p, 2048LL*1024);
  k_cvt<<<ew(1024LL*2048), tB, 0, stream>>>(prW1, pW1p, 1024LL*2048);
  k_cvt<<<ew(1024LL*1024), tB, 0, stream>>>(prW2, pW2p, 1024LL*1024);

  // --- inner forward ---
  gemm_wmma<true ><<<dim3(4096/128, (unsigned)(MT/128), 1), tB, 0, stream>>>(x_bf, 1024, 0, W1t, 1024, 0, Hb, 4096, 0, inb1, 0, 1024);
  k_swiglu<<<ew(MT*2048), tB, 0, stream>>>(Hb, sb, 2048, MT*2048);
  gemm_wmma<true ><<<dim3(1024/128, (unsigned)(MT/128), 1), tB, 0, stream>>>(sb, 2048, 0, W2t, 2048, 0, hb, 1024, 0, inb2, 0, 2048);
  gemm_wmma<true ><<<dim3(2048/128, (unsigned)(MT/128), 1), tB, 0, stream>>>(hb, 1024, 0, pW1t, 1024, 0, Hpb, 2048, 0, prb1, 0, 1024);
  k_swiglu<<<ew(MT*1024), tB, 0, stream>>>(Hpb, s2b, 1024, MT*1024);
  gemm_wmma<false><<<dim3(1024/128, (unsigned)(MT/128), 1), tB, 0, stream>>>(s2b, 1024, 0, pW2t, 1024, 0, prf, 1024, 0, prb2, 0, 1024);

  // --- loss gradient ---
  k_lossgrad<<<dim3((unsigned)MT), tB, 0, stream>>>(prf, tg, dprb);

  // --- backward ---
  gemm_wmma<true ><<<dim3(1024/128, (unsigned)(MT/128), 1), tB, 0, stream>>>(dprb, 1024, 0, pW2p, 1024, 0, ds2b, 1024, 0, nullptr, 0, 1024);
  k_dswiglu<<<ew(MT*1024), tB, 0, stream>>>(ds2b, Hpb, dHpb, 1024, MT*1024);
  gemm_wmma<true ><<<dim3(1024/128, (unsigned)(MT/128), 1), tB, 0, stream>>>(dHpb, 2048, 0, pW1p, 2048, 0, dhb, 1024, 0, nullptr, 0, 2048);
  gemm_wmma<true ><<<dim3(2048/128, (unsigned)(MT/128), 1), tB, 0, stream>>>(dhb, 1024, 0, W2p, 1024, 0, dsb, 2048, 0, nullptr, 0, 1024);
  k_dswiglu<<<ew(MT*2048), tB, 0, stream>>>(dsb, Hb, dHb, 2048, MT*2048);

  // weight grads: transpose grads/activations to K-contiguous layout, then TN GEMM
  k_transpose<u16><<<dim3(4096/32, 2048/32, Bd), trB, 0, stream>>>(dHb, (long long)Nd*4096, dHbT, 4096LL*2048, Nd, 4096);
  k_transpose<u16><<<dim3(1024/32, 2048/32, Bd), trB, 0, stream>>>(dhb, (long long)Nd*1024, dhbT, 1024LL*2048, Nd, 1024);
  k_transpose<u16><<<dim3(2048/32, 2048/32, Bd), trB, 0, stream>>>(sb, (long long)Nd*2048, sTb, 2048LL*2048, Nd, 2048);
  gemm_wmma<false><<<dim3(4096/128, 1024/128, Bd), tB, 0, stream>>>(xT, 2048, 1024LL*2048, dHbT, 2048, 4096LL*2048, gW1, 4096, 1024LL*4096, nullptr, 0, 2048);
  gemm_wmma<false><<<dim3(1024/128, 2048/128, Bd), tB, 0, stream>>>(sTb, 2048, 2048LL*2048, dhbT, 2048, 1024LL*2048, gW2, 1024, 2048LL*1024, nullptr, 0, 2048);
  k_colsum<<<dim3(4096/256, Bd), tB, 0, stream>>>(dHb, gb1, Nd, 4096);
  k_colsum<<<dim3(1024/256, Bd), tB, 0, stream>>>(dhb, gb2, Nd, 1024);

  // --- fast weights (u - |lr| * g), fused transpose to bf16 ---
  k_fastw<<<dim3(4096/32, 1024/32, Bd), trB, 0, stream>>>(uW1, gW1, lr, W1pt, 1024, 4096);
  k_fastw<<<dim3(1024/32, 2048/32, Bd), trB, 0, stream>>>(uW2, gW2, lr, W2pt, 2048, 1024);
  k_fastb<<<ew((long long)Bd*4096), tB, 0, stream>>>(ub1, gb1, lr, b1p, 4096, (long long)Bd*4096);
  k_fastb<<<ew((long long)Bd*1024), tB, 0, stream>>>(ub2, gb2, lr, b2p, 1024, (long long)Bd*1024);

  // --- fast-weight forward (batched: per-sample B matrices and biases) ---
  gemm_wmma<true ><<<dim3(4096/128, 2048/128, Bd), tB, 0, stream>>>(x_bf, 1024, (long long)Nd*1024, W1pt, 1024, 4096LL*1024, Hb, 4096, (long long)Nd*4096, b1p, 4096, 1024);
  k_swiglu<<<ew(MT*2048), tB, 0, stream>>>(Hb, sb, 2048, MT*2048);
  gemm_wmma<false><<<dim3(1024/128, 2048/128, Bd), tB, 0, stream>>>(sb, 2048, (long long)Nd*2048, W2pt, 2048, 1024LL*2048, prf, 1024, (long long)Nd*1024, b2p, 1024, 2048);

  // --- layernorm -> output ---
  k_layernorm<<<dim3((unsigned)MT), tB, 0, stream>>>(prf, lng, lnb, (float*)d_out);
}
